// BreakupConv_56427280335478
// MI455X (gfx1250) — matrix operand order repaired and checked by reading, not compiled, not run on hardware
//
#include <hip/hip_runtime.h>
#include <hip/hip_bf16.h>

// ---------------------------------------------------------------------------
// 3x3 VALID conv as implicit GEMM on CDNA5 (gfx1250, wave32, WMMA).
//
// GEMM view: Out[co][n] = sum_k W[co][k] * Im2col[k][n],
//   M = Cout = 128, N = 16*126*126 pixels, K = Cin*9 = 576.
//
// Precision: split-bf16 ("bf16x3") — v = hi(bf16) + lo(bf16 of residual);
// accumulate hi*hi + hi*lo + lo*hi into f32 via v_wmma_f32_16x16x32_bf16.
// ~f32 accuracy at ~2.7x the native V_WMMA_F32_16X16X4_F32 rate.
//
// Wave tiling: 8 waves as 4(M) x 2(N); each wave owns 32 Cout x 32 px
// (2x2 WMMA tiles) -> B-from-LDS traffic halved vs 1x4 tiling, extra A
// fragments come from the L2-resident (294 KB) weight planes instead.
// ---------------------------------------------------------------------------

typedef __attribute__((ext_vector_type(16))) __bf16 v16bf;
typedef __attribute__((ext_vector_type(8)))  __bf16 v8bf;
typedef __attribute__((ext_vector_type(8)))  float  v8f;

#define CIN   64
#define COUT  128
#define HIN   128
#define WIN   128
#define HOUT  126
#define WOUT  126
#define TW    64      // output-pixel tile width per block
#define TCOL  66      // input columns staged (TW + 2 halo)

union FragU { v16bf v; v8bf h[2]; };

// ---- weight prep: f32 [co][c][tap] -> bf16 hi/lo planes [tap][co][c] -------
__global__ __launch_bounds__(256) void prep_weights(
    const float* __restrict__ w, __bf16* __restrict__ whi, __bf16* __restrict__ wlo)
{
    int i = blockIdx.x * 256 + threadIdx.x;           // over COUT*CIN*9
    if (i >= COUT * CIN * 9) return;
    int j  = i % 9;
    int c  = (i / 9) % CIN;
    int co = i / (9 * CIN);
    float v = w[i];
    __bf16 hi = (__bf16)v;
    __bf16 lo = (__bf16)(v - (float)hi);
    int o = (j * COUT + co) * CIN + c;
    whi[o] = hi;
    wlo[o] = lo;
}

// ---- main conv kernel ------------------------------------------------------
__global__ __launch_bounds__(256) void conv3x3_wmma_bf16x3(
    const float* __restrict__ x,
    const __bf16* __restrict__ whi,
    const __bf16* __restrict__ wlo,
    float* __restrict__ out)
{
    // LDS: input window, channel-contiguous: s[(r*TCOL + w)*CIN + c]
    __shared__ __bf16 shi[3 * TCOL * CIN];   // 25344 B
    __shared__ __bf16 slo[3 * TCOL * CIN];   // 25344 B  (total ~50 KB of 320 KB)

    const int w0  = blockIdx.x * TW;   // 0 or 64
    const int h   = blockIdx.y;        // 0..125
    const int b   = blockIdx.z;        // 0..15
    const int tid = threadIdx.x;

    // ---- stage + split input tile: rows h..h+2, cols w0..w0+65, all 64 ch --
    const int NEL = 3 * TCOL * CIN;    // 12672 elements
    for (int e = tid; e < NEL; e += 256) {
        int wcol = e % TCOL;                 // fastest -> coalesced global reads
        int c    = (e / TCOL) % CIN;
        int r    = e / (TCOL * CIN);
        int wi   = w0 + wcol;
        if (wi > WIN - 1) wi = WIN - 1;      // clamp halo of edge tile (masked at store)
        float v  = x[((b * CIN + c) * HIN + (h + r)) * WIN + wi];
        __bf16 hi = (__bf16)v;
        __bf16 lo = (__bf16)(v - (float)hi);
        int o = (r * TCOL + wcol) * CIN + c;
        shi[o] = hi;
        slo[o] = lo;
    }
    __syncthreads();

    const int lane  = tid & 31;
    const int wave  = tid >> 5;
    const int mrow  = wave >> 1;         // 0..3 : 32 Cout rows each
    const int ncol  = wave & 1;          // 0..1 : 32 pixels each
    const int l15   = lane & 15;
    const int half  = lane >> 4;         // lane group 0/1

    // A-fragment K offsets (16-bit A 16x32 layout): lanes<16: K {0-7,16-23};
    // lanes>=16: K {8-15,24-31}. B (32x16): same K split as A.
    const int a_koff = half * 8;
    const int b_koff = half * 8;

    v8f acc[2][2] = {};                  // [mt][nt] : 32 Cout x 32 px per wave

    #pragma unroll
    for (int j = 0; j < 9; ++j) {
        const int dh = j / 3, dw = j % 3;
        #pragma unroll
        for (int ck = 0; ck < CIN; ck += 32) {
            // A fragments from global (weights are L2-resident, 294 KB total)
            FragU Ahi[2], Alo[2];
            #pragma unroll
            for (int mt = 0; mt < 2; ++mt) {
                const int co = mrow * 32 + mt * 16 + l15;   // A-matrix M = lane&15
                const __bf16* ah = &whi[(j * COUT + co) * CIN + ck + a_koff];
                const __bf16* al = &wlo[(j * COUT + co) * CIN + ck + a_koff];
                Ahi[mt].h[0] = *(const v8bf*)(ah);
                Ahi[mt].h[1] = *(const v8bf*)(ah + 16);
                Alo[mt].h[0] = *(const v8bf*)(al);
                Alo[mt].h[1] = *(const v8bf*)(al + 16);
            }

            #pragma unroll
            for (int nt = 0; nt < 2; ++nt) {
                const int col = ncol * 32 + nt * 16 + l15 + dw;  // input column
                const int bo  = (dh * TCOL + col) * CIN + ck + b_koff;
                FragU Bhi, Blo;
                Bhi.h[0] = *(const v8bf*)(&shi[bo]);             // ds_load_b128
                Bhi.h[1] = *(const v8bf*)(&shi[bo + 16]);
                Blo.h[0] = *(const v8bf*)(&slo[bo]);
                Blo.h[1] = *(const v8bf*)(&slo[bo + 16]);

                #pragma unroll
                for (int mt = 0; mt < 2; ++mt) {
                    acc[mt][nt] = __builtin_amdgcn_wmma_f32_16x16x32_bf16(
                        false, Ahi[mt].v, false, Bhi.v, (short)0, acc[mt][nt], false, false);
                    acc[mt][nt] = __builtin_amdgcn_wmma_f32_16x16x32_bf16(
                        false, Ahi[mt].v, false, Blo.v, (short)0, acc[mt][nt], false, false);
                    acc[mt][nt] = __builtin_amdgcn_wmma_f32_16x16x32_bf16(
                        false, Alo[mt].v, false, Bhi.v, (short)0, acc[mt][nt], false, false);
                }
            }
        }
    }

    // ---- store D: VGPR r holds M = r + (lane<16 ? 0 : 8), N = lane&15 ------
    #pragma unroll
    for (int nt = 0; nt < 2; ++nt) {
        const int wo = w0 + ncol * 32 + nt * 16 + l15;
        if (wo < WOUT) {
            #pragma unroll
            for (int mt = 0; mt < 2; ++mt) {
                const int mbase = mrow * 32 + mt * 16 + half * 8;
                #pragma unroll
                for (int r = 0; r < 8; ++r) {
                    const int co = mbase + r;
                    out[((b * COUT + co) * HOUT + h) * WOUT + wo] = acc[mt][nt][r];
                }
            }
        }
    }
}

// ---------------------------------------------------------------------------
extern "C" void kernel_launch(void* const* d_in, const int* in_sizes, int n_in,
                              void* d_out, int out_size, void* d_ws, size_t ws_size,
                              hipStream_t stream) {
    const float* x = (const float*)d_in[0];        // (16, 64, 128, 128) f32
    const float* w = (const float*)d_in[1];        // (128, 64, 9) f32
    float* out = (float*)d_out;                    // (16, 128, 126, 126) f32

    // workspace: hi/lo bf16 weight planes, [tap][cout][cin]  (288 KB total)
    const int NW = COUT * CIN * 9;                 // 73728
    __bf16* whi = (__bf16*)d_ws;
    __bf16* wlo = whi + NW;

    prep_weights<<<(NW + 255) / 256, 256, 0, stream>>>(w, whi, wlo);

    dim3 grid((WOUT + TW - 1) / TW, HOUT, 16);     // (2, 126, 16)
    conv3x3_wmma_bf16x3<<<grid, 256, 0, stream>>>(x, whi, wlo, out);
}
